// CausalSelfAttention_72662256714527
// MI455X (gfx1250) — matrix operand order repaired and checked by reading, not compiled, not run on hardware
//
#include <hip/hip_runtime.h>
#include <hip/hip_bf16.h>

typedef _Float16 half_t;
typedef __attribute__((ext_vector_type(16))) _Float16 v16h;
typedef __attribute__((ext_vector_type(8)))  _Float16 v8h;
typedef __attribute__((ext_vector_type(8)))  float    v8f;

namespace {
constexpr int kB = 8, kT = 1500, kC = 256, kH = 8, kD = 32;
constexpr int kM = kB * kT;          // 12000 rows (divisible by 16)
constexpr int kTp = 1536;            // padded T: Vt rows 32B-aligned, covers key idx < 1536
constexpr int kRowsPad = kM + 64;    // pad so b=7 edge-tile Q reads stay in bounds
constexpr size_t kRSZ  = (size_t)kRowsPad * kC;       // halves per [M,C] buffer
constexpr size_t kVTSZ = (size_t)kB * kH * kD * kTp;  // Vt halves
constexpr size_t kWSZ  = (size_t)kC * kC;             // weight halves
}

__global__ void cvt_f32_f16(const float* __restrict__ src, half_t* __restrict__ dst, int n) {
  int i = blockIdx.x * blockDim.x + threadIdx.x;
  if (i < n) dst[i] = (half_t)src[i];
}

// ---------------------------------------------------------------------------
// Transposed GEMM: computes Y^T tiles of Y = X @ W^T + b.
//   A = W subtile (16 n-rows x 32 c), B = X^T (32 c x 16 m-cols).
//   C-layout: lane = output row m, slots = 8 consecutive n -> coalesced stores.
// MODE 0: f16 row-major [M,256].  MODE 2: f32 row-major [M,256].
template <int MODE>
__global__ __launch_bounds__(128) void gemm_t(
    const half_t* __restrict__ X, const half_t* __restrict__ W,
    const float* __restrict__ bias, void* __restrict__ out) {
  const int lane = threadIdx.x & 31;
  const int wave = threadIdx.x >> 5;
  const int r    = lane & 15;
  const int hi   = lane >> 4;
  const int m0   = blockIdx.x * 16;
  const int n0   = wave * 64;

  v8f acc[4] = {};

  for (int kt = 0; kt < kC; kt += 32) {
    // B operand: column m == row m of X (contiguous 16 halves).
    v16h Bx = *(const v16h*)(X + (size_t)(m0 + r) * kC + kt + hi * 16);
#pragma unroll
    for (int s = 0; s < 4; ++s) {
      const half_t* wp = W + (size_t)(n0 + s * 16 + r) * kC + kt;
      v8h lo = *(const v8h*)(wp + hi * 16);
      v8h hh = *(const v8h*)(wp + hi * 16 + 8);
      v16h Aw;
#pragma unroll
      for (int j = 0; j < 8; ++j) { Aw[j] = lo[j]; Aw[8 + j] = hh[j]; }
      acc[s] = __builtin_amdgcn_wmma_f32_16x16x32_f16(
          false, Aw, false, Bx, (short)0, acc[s], false, false);
    }
  }

  const int m = m0 + r;
#pragma unroll
  for (int s = 0; s < 4; ++s) {
    const int nb = n0 + s * 16 + 8 * hi;            // 8 consecutive n per lane
    v8f bv = *(const v8f*)(bias + nb);
    v8f v  = acc[s] + bv;
    if (MODE == 0) {
      v8h hv;
#pragma unroll
      for (int i = 0; i < 8; ++i) hv[i] = (half_t)v[i];
      *(v8h*)((half_t*)out + (size_t)m * kC + nb) = hv;
    } else {
      *(v8f*)((float*)out + (size_t)m * kC + nb) = v;
    }
  }
}

// Normal-orientation GEMM producing per-head transposed V: Vt[B,H,D,Tp].
// C-layout: lane = n (fixed head/d), slots = m rows -> t varies per slot.
__global__ __launch_bounds__(128) void gemm_vt(
    const half_t* __restrict__ X, const half_t* __restrict__ W,
    const float* __restrict__ bias, half_t* __restrict__ Vt) {
  const int lane = threadIdx.x & 31;
  const int wave = threadIdx.x >> 5;
  const int r    = lane & 15;
  const int hi   = lane >> 4;
  const int m0   = blockIdx.x * 16;
  const int n0   = wave * 64;

  v8f acc[4] = {};

  for (int kt = 0; kt < kC; kt += 32) {
    const half_t* ap = X + (size_t)(m0 + r) * kC + kt;
    v8h alo = *(const v8h*)(ap + hi * 16);
    v8h ahi = *(const v8h*)(ap + hi * 16 + 8);
    v16h A;
#pragma unroll
    for (int j = 0; j < 8; ++j) { A[j] = alo[j]; A[8 + j] = ahi[j]; }
#pragma unroll
    for (int s = 0; s < 4; ++s) {
      const int n = n0 + s * 16 + r;
      v16h Bv = *(const v16h*)(W + (size_t)n * kC + kt + hi * 16);
      acc[s] = __builtin_amdgcn_wmma_f32_16x16x32_f16(
          false, A, false, Bv, (short)0, acc[s], false, false);
    }
  }

#pragma unroll
  for (int s = 0; s < 4; ++s) {
    const int n  = n0 + s * 16 + r;
    const int hh = n >> 5, d = n & 31;
    const float bn = bias[n];
#pragma unroll
    for (int i = 0; i < 8; ++i) {
      const int mm = m0 + i + 8 * hi;
      const int bb = mm / kT, t = mm % kT;
      Vt[(((size_t)bb * kH + hh) * kD + d) * kTp + t] = (half_t)(acc[s][i] + bn);
    }
  }
}

// ---------------------------------------------------------------------------
// Flash attention, transposed orientation. One wave per (b, h, 16-query tile).
//   S^T = K * Q^T : lane = query column, slots = key rows.
//   O^T = V^T * P^T accumulated in f32; keys streamed 64 at a time.
__global__ __launch_bounds__(32) void attn_fa(
    const half_t* __restrict__ Q, const half_t* __restrict__ K,
    const half_t* __restrict__ Vt, half_t* __restrict__ Y) {
  const int lane = threadIdx.x;
  const int r = lane & 15, hi = lane >> 4;
  const int q0 = blockIdx.x * 16;
  const int b = blockIdx.y >> 3, h = blockIdx.y & 7;
  const int qg = q0 + r;                 // this lane's query
  const float NEG = -1e30f;
  const float qscale = 0.25503472f;      // (1/sqrt(32)) * log2(e): softmax in exp2 domain

  // B operand: Q^T (32 d x 16 queries); lane's column is a contiguous Q row.
  v16h Bq;
  {
    const int qc = qg < kT ? qg : (kT - 1);
    v16h t = *(const v16h*)(Q + (size_t)(b * kT + qc) * kC + h * kD + hi * 16);
#pragma unroll
    for (int j = 0; j < 16; ++j) Bq[j] = (half_t)((float)t[j] * qscale);
  }

  float m = NEG, l = 0.f;
  v8f o0 = {}, o1 = {};
  const half_t* kbase = K + (size_t)(b * kT) * kC + h * kD;
  const half_t* vb    = Vt + ((size_t)(b * kH + h) * kD) * kTp;

  const int kend = q0 + 16;
  for (int k0 = 0; k0 < kend; k0 += 64) {
    const bool full = (k0 + 63 <= q0) && (k0 + 63 < kT);  // uniform branch
    v8f S[4];
#pragma unroll
    for (int s = 0; s < 4; ++s) {
      // A operand: K subtile (16 keys x 32 d); lane r = key row.
      const int kr  = k0 + s * 16 + r;
      const int krc = kr < kT ? kr : (kT - 1);
      const half_t* kp = kbase + (size_t)krc * kC;
      v8h lo = *(const v8h*)(kp + hi * 16);
      v8h hh = *(const v8h*)(kp + hi * 16 + 8);
      v16h Ak;
#pragma unroll
      for (int j = 0; j < 8; ++j) { Ak[j] = lo[j]; Ak[8 + j] = hh[j]; }
      v8f z = {};
      S[s] = __builtin_amdgcn_wmma_f32_16x16x32_f16(
          false, Ak, false, Bq, (short)0, z, false, false);
      if (!full) {
#pragma unroll
        for (int i = 0; i < 8; ++i) {
          const int kg = k0 + s * 16 + i + 8 * hi;   // slot's key index
          S[s][i] = (kg <= qg && kg < kT) ? S[s][i] : NEG;
        }
      }
    }

    // Online softmax: in-lane reduction over 32 keys + one xor-16 shuffle.
    float mx = NEG;
#pragma unroll
    for (int s = 0; s < 4; ++s)
#pragma unroll
      for (int i = 0; i < 8; ++i) mx = fmaxf(mx, S[s][i]);
    mx = fmaxf(mx, __shfl_xor(mx, 16, 32));
    const float mn = fmaxf(m, mx);
    const float rescale = exp2f(m - mn);
    float rs = 0.f;
#pragma unroll
    for (int s = 0; s < 4; ++s)
#pragma unroll
      for (int i = 0; i < 8; ++i) {
        const float p = exp2f(S[s][i] - mn);
        S[s][i] = p;
        rs += p;
      }
    rs += __shfl_xor(rs, 16, 32);
    l = l * rescale + rs;
    m = mn;
#pragma unroll
    for (int i = 0; i < 8; ++i) { o0[i] *= rescale; o1[i] *= rescale; }

    // Pack P^T into f16 pairs (lane-local) for the B operand.
    unsigned int u[4][4];
#pragma unroll
    for (int s = 0; s < 4; ++s)
#pragma unroll
      for (int j = 0; j < 4; ++j) {
        union { half_t hx[2]; unsigned int w; } cv;
        cv.hx[0] = (half_t)S[s][2 * j];
        cv.hx[1] = (half_t)S[s][2 * j + 1];
        u[s][j] = cv.w;
      }

#pragma unroll
    for (int g = 0; g < 2; ++g) {
      // Build P^T B operand (32 keys x 16 queries) via one half-wave exchange.
      union { unsigned int w[8]; v16h h; } Bp;
#pragma unroll
      for (int j = 0; j < 4; ++j) {
        const unsigned int snd = hi ? u[2 * g][j]     : u[2 * g + 1][j];
        const unsigned int kep = hi ? u[2 * g + 1][j] : u[2 * g][j];
        const unsigned int rcv = __shfl_xor(snd, 16, 32);
        Bp.w[j]     = hi ? rcv : kep;   // keys +0..7  (lanes 0-15) / +16..23 (16-31)
        Bp.w[4 + j] = hi ? kep : rcv;   // keys +8..15 (lanes 0-15) / +24..31 (16-31)
      }
      // A operands: V^T tiles (16 d x 32 keys), contiguous along t.
#pragma unroll
      for (int dt = 0; dt < 2; ++dt) {
        const half_t* vp = vb + (size_t)(dt * 16 + r) * kTp + k0 + g * 32;
        v8h lo = *(const v8h*)(vp + hi * 16);
        v8h hh = *(const v8h*)(vp + hi * 16 + 8);
        v16h Av;
#pragma unroll
        for (int j = 0; j < 8; ++j) { Av[j] = lo[j]; Av[8 + j] = hh[j]; }
        if (dt == 0)
          o0 = __builtin_amdgcn_wmma_f32_16x16x32_f16(false, Av, false, Bp.h, (short)0, o0, false, false);
        else
          o1 = __builtin_amdgcn_wmma_f32_16x16x32_f16(false, Av, false, Bp.h, (short)0, o1, false, false);
      }
    }
  }

  // Epilogue: lane owns one query row; d is contiguous across slots -> b128 stores.
  if (qg < kT) {
    const float inv = 1.0f / l;
    half_t* yp = Y + (size_t)(b * kT + qg) * kC + h * kD;
    v8h y0, y1;
#pragma unroll
    for (int i = 0; i < 8; ++i) {
      y0[i] = (half_t)(o0[i] * inv);
      y1[i] = (half_t)(o1[i] * inv);
    }
    *(v8h*)(yp + 8 * hi)      = y0;   // d = 0..7 / 8..15
    *(v8h*)(yp + 16 + 8 * hi) = y1;   // d = 16..23 / 24..31
  }
}

extern "C" void kernel_launch(void* const* d_in, const int* in_sizes, int n_in,
                              void* d_out, int out_size, void* d_ws, size_t ws_size,
                              hipStream_t stream) {
  (void)in_sizes; (void)n_in; (void)out_size; (void)ws_size;
  const float* x  = (const float*)d_in[0];
  const float* Wq = (const float*)d_in[1];
  const float* bq = (const float*)d_in[2];
  const float* Wk = (const float*)d_in[3];
  const float* bk = (const float*)d_in[4];
  const float* Wv = (const float*)d_in[5];
  const float* bv = (const float*)d_in[6];
  const float* Wp = (const float*)d_in[7];
  const float* bp = (const float*)d_in[8];

  half_t* ws   = (half_t*)d_ws;
  half_t* X16  = ws;                 // [RowsPad, C]
  half_t* Q16  = X16 + kRSZ;
  half_t* K16  = Q16 + kRSZ;
  half_t* Y16  = K16 + kRSZ;
  half_t* Vt   = Y16 + kRSZ;         // [B,H,D,Tp]
  half_t* Wq16 = Vt + kVTSZ;
  half_t* Wk16 = Wq16 + kWSZ;
  half_t* Wv16 = Wk16 + kWSZ;
  half_t* Wp16 = Wv16 + kWSZ;

  const int nx = kM * kC;
  cvt_f32_f16<<<(nx + 255) / 256, 256, 0, stream>>>(x, X16, nx);
  cvt_f32_f16<<<((int)kWSZ + 255) / 256, 256, 0, stream>>>(Wq, Wq16, (int)kWSZ);
  cvt_f32_f16<<<((int)kWSZ + 255) / 256, 256, 0, stream>>>(Wk, Wk16, (int)kWSZ);
  cvt_f32_f16<<<((int)kWSZ + 255) / 256, 256, 0, stream>>>(Wv, Wv16, (int)kWSZ);
  cvt_f32_f16<<<((int)kWSZ + 255) / 256, 256, 0, stream>>>(Wp, Wp16, (int)kWSZ);

  dim3 gg(kM / 16);  // 750 M-tiles, 4 waves x 64 output cols each
  gemm_t<0><<<gg, 128, 0, stream>>>(X16, Wq16, bq, Q16);
  gemm_t<0><<<gg, 128, 0, stream>>>(X16, Wk16, bk, K16);
  gemm_vt  <<<gg, 128, 0, stream>>>(X16, Wv16, bv, Vt);

  dim3 ga((kT + 15) / 16, kB * kH);  // 94 x 64 wave-tiles
  attn_fa<<<ga, 32, 0, stream>>>(Q16, K16, Vt, Y16);

  gemm_t<2><<<gg, 128, 0, stream>>>(Y16, Wp16, bp, d_out);
}
